// SwinTransformerBlock_14929306321541
// MI455X (gfx1250) — matrix operand order repaired and checked by reading, not compiled, not run on hardware
//
#include <hip/hip_runtime.h>
#include <math.h>

// ---------------- CDNA5 WMMA types ----------------
typedef __attribute__((ext_vector_type(16))) __bf16 v16bf;
typedef __attribute__((ext_vector_type(8)))  float  v8f;

union Frag { v16bf v; unsigned int u[8]; };

static constexpr int NTOK = 64 * 56 * 56;          // 200704, divisible by 128
static constexpr float ATT_SCALE = 0.17677669529663687f; // 32^-0.5

__device__ __forceinline__ unsigned short f2bf(float f) {
  unsigned int x = __float_as_uint(f);
  return (unsigned short)((x + 0x7FFFu + ((x >> 16) & 1u)) >> 16);
}

// A-matrix (16x32 bf16) per-lane K pair base for VGPR r, lane half = lane>>4
__device__ __forceinline__ int a_kpair(int r, int khalf) {
  return 2 * r + 16 * khalf;
}

__device__ __forceinline__ v8f wmma_bf16(const Frag& a, const Frag& b, v8f c) {
  return __builtin_amdgcn_wmma_f32_16x16x32_bf16(false, a.v, false, b.v,
                                                 (short)0, c, false, false);
}

// gfx1250 async global->LDS copy (ASYNCcnt-tracked), 16B per lane
__device__ __forceinline__ void async_ld_b128(unsigned int lds_off, const void* g) {
  asm volatile("global_load_async_to_lds_b128 %0, %1, off"
               :: "v"(lds_off), "v"(g) : "memory");
}
__device__ __forceinline__ void wait_async0() {
  asm volatile("s_wait_asynccnt 0x0" ::: "memory");
}
// flat addr of a shared object: low 32 bits are the LDS byte offset
__device__ __forceinline__ unsigned int lds_off(const void* p) {
  return (unsigned int)(size_t)p;
}

// window-order row -> original token index (inverse of shift+partition)
__device__ __forceinline__ size_t win_to_orig(int r) {
  int win = r / 49, n = r - win * 49;
  int b = win >> 6, wi = win & 63;
  int wy = wi >> 3, wx = wi & 7;
  int ny = n / 7, nx = n - ny * 7;
  int hh = wy * 7 + ny + 3; if (hh >= 56) hh -= 56;
  int ww = wx * 7 + nx + 3; if (ww >= 56) ww -= 56;
  return (size_t)b * 3136 + (size_t)hh * 56 + (size_t)ww;
}

// ---------------- fp32 -> bf16 weight convert ----------------
__global__ void cvt_bf16_kernel(const float* __restrict__ src,
                                unsigned short* __restrict__ dst, int n) {
  int i = blockIdx.x * 256 + threadIdx.x;
  if (i < n) dst[i] = f2bf(src[i]);
}

// ---------------- LayerNorm (one wave per token, C=128) ----------------
__global__ __launch_bounds__(256) void ln_kernel(
    const float* __restrict__ x, const float* __restrict__ g,
    const float* __restrict__ bt, unsigned short* __restrict__ out, int perm) {
  int row  = blockIdx.x * 8 + (threadIdx.x >> 5);
  int lane = threadIdx.x & 31;
  size_t src = perm ? win_to_orig(row) : (size_t)row;
  int c0 = lane * 4;
  const float4 v = *(const float4*)(x + src * 128 + c0);
  float s  = v.x + v.y + v.z + v.w;
  float s2 = v.x * v.x + v.y * v.y + v.z * v.z + v.w * v.w;
  #pragma unroll
  for (int m = 1; m < 32; m <<= 1) {
    s  += __shfl_xor(s,  m, 32);
    s2 += __shfl_xor(s2, m, 32);
  }
  float mean = s * (1.0f / 128.0f);
  float var  = s2 * (1.0f / 128.0f) - mean * mean;
  float inv  = rsqrtf(var + 1e-5f);
  float4 gg = *(const float4*)(g  + c0);
  float4 bb = *(const float4*)(bt + c0);
  float y0 = (v.x - mean) * inv * gg.x + bb.x;
  float y1 = (v.y - mean) * inv * gg.y + bb.y;
  float y2 = (v.z - mean) * inv * gg.z + bb.z;
  float y3 = (v.w - mean) * inv * gg.w + bb.w;
  unsigned int lo = (unsigned int)f2bf(y0) | ((unsigned int)f2bf(y1) << 16);
  unsigned int hi = (unsigned int)f2bf(y2) | ((unsigned int)f2bf(y3) << 16);
  *(uint2*)(out + (size_t)row * 128 + c0) = make_uint2(lo, hi);
}

// ---------------- generic bf16 WMMA GEMM ----------------
// block tile 128x64, 8 waves of 32x32; A staged via async global->LDS with
// double buffering; B transposed through VGPRs into padded LDS.
// mode 0: bf16 out   mode 1: GELU->bf16
// mode 2: fp32 resid[perm]+val (window-reverse scatter)   mode 3: fp32 resid+val
__global__ __launch_bounds__(256) void gemm_kernel(
    const unsigned short* __restrict__ A, const unsigned short* __restrict__ Bw,
    const float* __restrict__ bias, unsigned short* __restrict__ outB,
    float* __restrict__ outF, const float* __restrict__ resid,
    int M, int N, int K, int mode) {
  __shared__ __align__(16) unsigned short As[2][128 * 32]; // [m][k]
  __shared__ __align__(16) unsigned short Bs[2][64 * 40];  // [n][k] pad->40
  int tid = threadIdx.x;
  int wave = tid >> 5, lane = tid & 31;
  int mlane = lane & 15, khalf = lane >> 4;
  int m0 = blockIdx.y * 128, n0 = blockIdx.x * 64;
  int wm = (wave & 3) * 32, wn = (wave >> 2) * 32;

  int arow = tid >> 1, ahalf = tid & 1;     // A staging: 16B per thread
  int bn = tid & 63, bkg = tid >> 6;        // B staging: 8 u16 per thread

  v8f zero = {0.f, 0.f, 0.f, 0.f, 0.f, 0.f, 0.f, 0.f};
  v8f acc[2][2] = {{zero, zero}, {zero, zero}};

  // prologue: stage chunk 0 into buffer 0
  async_ld_b128(lds_off(&As[0][arow * 32 + ahalf * 16]),
                A + (size_t)(m0 + arow) * K + ahalf * 16);
  #pragma unroll
  for (int k8 = 0; k8 < 8; k8++) {
    int k = bkg * 8 + k8;
    Bs[0][bn * 40 + k] = Bw[(size_t)k * N + n0 + bn];
  }

  int buf = 0;
  for (int kk = 0; kk < K; kk += 32) {
    wait_async0();
    __syncthreads();

    if (kk + 32 < K) { // stage next chunk into the other buffer
      async_ld_b128(lds_off(&As[buf ^ 1][arow * 32 + ahalf * 16]),
                    A + (size_t)(m0 + arow) * K + kk + 32 + ahalf * 16);
      #pragma unroll
      for (int k8 = 0; k8 < 8; k8++) {
        int k = bkg * 8 + k8;
        Bs[buf ^ 1][bn * 40 + k] = Bw[(size_t)(kk + 32 + k) * N + n0 + bn];
      }
    }

    Frag afr[2], bfr[2];
    #pragma unroll
    for (int i = 0; i < 2; i++) {
      const unsigned int* ap =
          (const unsigned int*)(&As[buf][(wm + i * 16 + mlane) * 32]);
      #pragma unroll
      for (int r = 0; r < 8; r++) afr[i].u[r] = ap[a_kpair(r, khalf) >> 1];
    }
    #pragma unroll
    for (int j = 0; j < 2; j++) {
      const unsigned int* bp =
          (const unsigned int*)(&Bs[buf][(wn + j * 16 + mlane) * 40]);
      #pragma unroll
      for (int r = 0; r < 8; r++) bfr[j].u[r] = bp[(2 * r + 16 * khalf) >> 1];
    }
    #pragma unroll
    for (int i = 0; i < 2; i++)
      #pragma unroll
      for (int j = 0; j < 2; j++) acc[i][j] = wmma_bf16(afr[i], bfr[j], acc[i][j]);
    __syncthreads();
    buf ^= 1;
  }

  // epilogue: C layout lane L -> N = L&15, M = r + 8*(L>>4)
  #pragma unroll
  for (int i = 0; i < 2; i++) {
    #pragma unroll
    for (int j = 0; j < 2; j++) {
      int ncol = n0 + wn + j * 16 + mlane;
      float bv = bias[ncol];
      #pragma unroll
      for (int r = 0; r < 8; r++) {
        int mrow = m0 + wm + i * 16 + r + 8 * khalf;
        float vv = acc[i][j][r] + bv;
        if (mode == 0) {
          outB[(size_t)mrow * N + ncol] = f2bf(vv);
        } else if (mode == 1) {
          float gl = 0.5f * vv * (1.0f + erff(vv * 0.70710678118f));
          outB[(size_t)mrow * N + ncol] = f2bf(gl);
        } else if (mode == 2) {
          size_t o = win_to_orig(mrow);
          outF[o * 128 + ncol] = resid[o * 128 + ncol] + vv;
        } else {
          outF[(size_t)mrow * 128 + ncol] = resid[(size_t)mrow * 128 + ncol] + vv;
        }
      }
    }
  }
}

// ---------------- windowed attention: one wave per (window, head) ----------
// Whole window qkv tile (49x384 bf16, contiguous 37632B) is async-copied to
// LDS once; S = q@k^T via WMMA, fused bias+mask softmax in-register,
// P -> LDS bf16, out = P@v via WMMA.
__global__ __launch_bounds__(128) void attn_kernel(
    const unsigned short* __restrict__ qkv, const float* __restrict__ relb,
    unsigned short* __restrict__ out) {
  __shared__ __align__(16) unsigned short T[49 * 384];  // window qkv tile
  __shared__ __align__(16) unsigned short P[4][64 * 64];
  int win = blockIdx.x;
  int h = threadIdx.x >> 5, lane = threadIdx.x & 31;
  int mlane = lane & 15, hl = lane >> 4;
  int wi = win & 63, wy = wi >> 3, wx = wi & 7;
  size_t rowbase = (size_t)win * 49;

  { // async stage: 2352 x 16B coalesced copies
    const char* g = (const char*)(qkv + rowbase * 384);
    unsigned int tb = lds_off(&T[0]);
    for (int it = 0; it < 19; ++it) {
      int idx = it * 128 + threadIdx.x;
      if (idx < 2352) async_ld_b128(tb + idx * 16, g + idx * 16);
    }
    wait_async0();
    __syncthreads();
  }

  // K-matrix B-frags for all 4 column tiles (K dim = head_dim = 32)
  Frag kf[4];
  #pragma unroll
  for (int nt = 0; nt < 4; nt++) {
    int j = nt * 16 + mlane;
    const unsigned short* kp = &T[(j < 49 ? j : 0) * 384 + 128 + h * 32];
    #pragma unroll
    for (int r = 0; r < 8; r++) {
      int d = 2 * r + 16 * hl;
      kf[nt].u[r] = (j < 49) ? *(const unsigned int*)(kp + d) : 0u;
    }
  }

  // per-lane column metadata (col = nt*16 + mlane)
  int yj[4], xj[4], cntj[4]; bool jval[4];
  #pragma unroll
  for (int nt = 0; nt < 4; nt++) {
    int cj = nt * 16 + mlane;
    jval[nt] = cj < 49;
    int y = cj / 7, x = cj - (cj / 7) * 7;
    yj[nt] = y; xj[nt] = x;
    int hs = wy * 7 + y, wsc = wx * 7 + x;
    cntj[nt] = (hs < 49 ? 0 : (hs < 53 ? 1 : 2)) * 3 +
               (wsc < 49 ? 0 : (wsc < 53 ? 1 : 2));
  }

  v8f zero = {0.f, 0.f, 0.f, 0.f, 0.f, 0.f, 0.f, 0.f};

  for (int mt = 0; mt < 4; mt++) {
    Frag qf;
    int qi = mt * 16 + mlane;
    const unsigned short* qp = &T[(qi < 49 ? qi : 0) * 384 + h * 32];
    #pragma unroll
    for (int r = 0; r < 8; r++)
      qf.u[r] = (qi < 49) ? *(const unsigned int*)(qp + a_kpair(r, hl)) : 0u;

    v8f s[4] = {zero, zero, zero, zero};
    #pragma unroll
    for (int nt = 0; nt < 4; nt++) s[nt] = wmma_bf16(qf, kf[nt], s[nt]);

    // fused scale + rel-pos bias + shift mask + row softmax
    #pragma unroll
    for (int r = 0; r < 8; r++) {
      int m = mt * 16 + r + 8 * hl;   // C layout row
      bool ival = m < 49;
      int yi = m / 7, xi = m - (m / 7) * 7;
      int hsi = wy * 7 + yi, wsi = wx * 7 + xi;
      int cnti = (hsi < 49 ? 0 : (hsi < 53 ? 1 : 2)) * 3 +
                 (wsi < 49 ? 0 : (wsi < 53 ? 1 : 2));
      float vals[4];
      #pragma unroll
      for (int nt = 0; nt < 4; nt++) {
        float sv = s[nt][r] * ATT_SCALE;
        if (ival && jval[nt]) {
          int ridx = (yi - yj[nt] + 6) * 13 + (xi - xj[nt] + 6);
          sv += relb[ridx * 4 + h];
          if (cnti != cntj[nt]) sv -= 100.0f;
        } else {
          sv = -1e30f;
        }
        vals[nt] = sv;
      }
      float mx = fmaxf(fmaxf(vals[0], vals[1]), fmaxf(vals[2], vals[3]));
      #pragma unroll
      for (int d2 = 1; d2 < 16; d2 <<= 1) mx = fmaxf(mx, __shfl_xor(mx, d2, 32));
      float e[4], sum = 0.0f;
      #pragma unroll
      for (int nt = 0; nt < 4; nt++) { e[nt] = __expf(vals[nt] - mx); sum += e[nt]; }
      #pragma unroll
      for (int d2 = 1; d2 < 16; d2 <<= 1) sum += __shfl_xor(sum, d2, 32);
      float inv = (sum > 0.0f) ? 1.0f / sum : 0.0f;
      #pragma unroll
      for (int nt = 0; nt < 4; nt++)
        P[h][m * 64 + nt * 16 + mlane] = f2bf(e[nt] * inv);
    }
  }

  // V B-frags from LDS tile: K dim = 64 key tokens (2 k-steps), N = 32 dims
  Frag vf[2][2];
  #pragma unroll
  for (int nt = 0; nt < 2; nt++) {
    int n = nt * 16 + mlane;
    #pragma unroll
    for (int ks = 0; ks < 2; ks++) {
      #pragma unroll
      for (int r = 0; r < 8; r++) {
        int j0 = ks * 32 + 2 * r + 16 * hl;
        unsigned int lo = (j0 < 49)
            ? (unsigned int)T[j0 * 384 + 256 + h * 32 + n] : 0u;
        unsigned int hi = (j0 + 1 < 49)
            ? (unsigned int)T[(j0 + 1) * 384 + 256 + h * 32 + n] : 0u;
        vf[nt][ks].u[r] = lo | (hi << 16);
      }
    }
  }

  for (int mt = 0; mt < 4; mt++) {
    Frag pf[2];
    #pragma unroll
    for (int ks = 0; ks < 2; ks++) {
      const unsigned int* pp =
          (const unsigned int*)(&P[h][0] + (mt * 16 + mlane) * 64 + ks * 32);
      #pragma unroll
      for (int r = 0; r < 8; r++) pf[ks].u[r] = pp[a_kpair(r, hl) >> 1];
    }
    v8f o2[2] = {zero, zero};
    #pragma unroll
    for (int nt = 0; nt < 2; nt++) {
      o2[nt] = wmma_bf16(pf[0], vf[nt][0], o2[nt]);
      o2[nt] = wmma_bf16(pf[1], vf[nt][1], o2[nt]);
    }
    #pragma unroll
    for (int nt = 0; nt < 2; nt++) {
      int ncol = h * 32 + nt * 16 + mlane;
      #pragma unroll
      for (int r = 0; r < 8; r++) {
        int m = mt * 16 + r + 8 * hl;
        if (m < 49) out[(rowbase + m) * 128 + ncol] = f2bf(o2[nt][r]);
      }
    }
  }
}

// ---------------- host launch ----------------
extern "C" void kernel_launch(void* const* d_in, const int* in_sizes, int n_in,
                              void* d_out, int out_size, void* d_ws, size_t ws_size,
                              hipStream_t stream) {
  (void)in_sizes; (void)n_in; (void)out_size; (void)ws_size;
  const float* x      = (const float*)d_in[0];
  const float* n1g    = (const float*)d_in[1];
  const float* n1b    = (const float*)d_in[2];
  const float* qkv_w  = (const float*)d_in[3];
  const float* qkv_b  = (const float*)d_in[4];
  const float* proj_w = (const float*)d_in[5];
  const float* proj_b = (const float*)d_in[6];
  const float* relb   = (const float*)d_in[7];
  const float* n2g    = (const float*)d_in[8];
  const float* n2b    = (const float*)d_in[9];
  const float* fc1_w  = (const float*)d_in[10];
  const float* fc1_b  = (const float*)d_in[11];
  const float* fc2_w  = (const float*)d_in[12];
  const float* fc2_b  = (const float*)d_in[13];
  float* outp = (float*)d_out;

  const int M = NTOK;
  char* ws = (char*)d_ws;
  size_t off = 0;
  auto alloc = [&](size_t bytes) -> void* {
    off = (off + 255) & ~(size_t)255;
    void* p = ws + off;
    off += bytes;
    return p;
  };

  unsigned short* wqkv = (unsigned short*)alloc((size_t)128 * 384 * 2);
  unsigned short* wprj = (unsigned short*)alloc((size_t)128 * 128 * 2);
  unsigned short* wfc1 = (unsigned short*)alloc((size_t)128 * 512 * 2);
  unsigned short* wfc2 = (unsigned short*)alloc((size_t)512 * 128 * 2);
  // region A: qkv [M,384] (dead after attn) + attn [M,128] in the tail;
  // f1 [M,512] reuses the whole region later. M*384 + M*128 == M*512.
  unsigned short* regA  = (unsigned short*)alloc((size_t)M * 512 * 2);
  unsigned short* qkvB  = regA;
  unsigned short* attnB = regA + (size_t)M * 384;
  unsigned short* f1    = regA;
  unsigned short* xw = (unsigned short*)alloc((size_t)M * 128 * 2); // also h2
  unsigned short* h2 = xw;
  float* x1 = (float*)alloc((size_t)M * 128 * 4);

  cvt_bf16_kernel<<<(128 * 384 + 255) / 256, 256, 0, stream>>>(qkv_w, wqkv, 128 * 384);
  cvt_bf16_kernel<<<(128 * 128 + 255) / 256, 256, 0, stream>>>(proj_w, wprj, 128 * 128);
  cvt_bf16_kernel<<<(128 * 512 + 255) / 256, 256, 0, stream>>>(fc1_w, wfc1, 128 * 512);
  cvt_bf16_kernel<<<(512 * 128 + 255) / 256, 256, 0, stream>>>(fc2_w, wfc2, 512 * 128);

  ln_kernel<<<M / 8, 256, 0, stream>>>(x, n1g, n1b, xw, 1);

  { dim3 g(384 / 64, M / 128);
    gemm_kernel<<<g, 256, 0, stream>>>(xw, wqkv, qkv_b, qkvB, nullptr, nullptr,
                                       M, 384, 128, 0); }

  attn_kernel<<<4096, 128, 0, stream>>>(qkvB, relb, attnB);

  { dim3 g(128 / 64, M / 128);
    gemm_kernel<<<g, 256, 0, stream>>>(attnB, wprj, proj_b, nullptr, x1, x,
                                       M, 128, 128, 2); }

  ln_kernel<<<M / 8, 256, 0, stream>>>(x1, n2g, n2b, h2, 0);

  { dim3 g(512 / 64, M / 128);
    gemm_kernel<<<g, 256, 0, stream>>>(h2, wfc1, fc1_b, f1, nullptr, nullptr,
                                       M, 512, 128, 1); }

  { dim3 g(128 / 64, M / 128);
    gemm_kernel<<<g, 256, 0, stream>>>(f1, wfc2, fc2_b, nullptr, outp, x1,
                                       M, 128, 512, 3); }
}
